// Net_41446434406715
// MI455X (gfx1250) — compile-verified, matrix operand order checked by reading, not run on hardware
//
#include <hip/hip_runtime.h>
#include <hip/hip_bf16.h>
#include <math.h>

typedef __attribute__((ext_vector_type(16))) __bf16 v16bf;
typedef __attribute__((ext_vector_type(8)))  float  v8f;

#define N_FEAT_C 500
#define N_CLS_C  7

// ---------------------------------------------------------------------------
// Degree / normalization kernels
// ---------------------------------------------------------------------------
__global__ __launch_bounds__(256) void k_init_deg(float* __restrict__ deg, int n) {
    int i = blockIdx.x * blockDim.x + threadIdx.x;
    if (i < n) deg[i] = 1.0f;  // self-loop contributes exactly 1 per node
}

__global__ __launch_bounds__(256) void k_count(const long long* __restrict__ dst,
                                               float* __restrict__ deg, int e_cnt) {
    int e = blockIdx.x * blockDim.x + threadIdx.x;
    if (e < e_cnt) atomicAdd(&deg[(int)dst[e]], 1.0f);
}

__global__ __launch_bounds__(256) void k_rsqrt_inplace(float* __restrict__ d, int n) {
    int i = blockIdx.x * blockDim.x + threadIdx.x;
    if (i < n) {
        float v = d[i];
        d[i] = (v > 0.0f) ? rsqrtf(v) : 0.0f;
    }
}

// ---------------------------------------------------------------------------
// Projection: y0[n, 0..7] = x[n, :] @ W^T  (col 7 padded with zeros)
// One wave32 per 16-node tile; v_wmma_f32_16x16x32_bf16 over K=512 (padded).
// ---------------------------------------------------------------------------
__global__ __launch_bounds__(256)
void sgc_project_wmma(const float* __restrict__ x, const float* __restrict__ w,
                      float* __restrict__ y, int n_nodes) {
    const int wave = (int)((blockIdx.x * blockDim.x + threadIdx.x) >> 5);
    const int lane = threadIdx.x & 31;
    const int n_tiles = (n_nodes + 15) / 16;
    if (wave >= n_tiles) return;              // wave-uniform exit

    const int node_base = wave * 16;
    const int half = lane >> 4;               // lane group 0/1
    const int lid  = lane & 15;

    // A operand: this lane supplies row m = lid of the node tile.
    int row_node = node_base + lid;
    if (row_node >= n_nodes) row_node = n_nodes - 1;   // clamp; not stored
    const float* __restrict__ xrow = x + (size_t)row_node * N_FEAT_C;

    // B operand: this lane supplies column n = lid (class). n >= 7 -> zeros.
    const bool  col_ok = (lid < N_CLS_C);
    const float cmask  = col_ok ? 1.0f : 0.0f;
    const float* __restrict__ wrow = w + (size_t)(col_ok ? lid : 0) * N_FEAT_C;

    v8f acc = {};
    for (int k0 = 0; k0 < N_FEAT_C; k0 += 32) {
        // K enumeration for this lane (matches 16-bit A layout, ISA 7.12.2):
        //   elems 0..7  -> K = k0 + half*8 + j
        //   elems 8..15 -> K = k0 + 16 + half*8 + j
        const int f0 = k0 + half * 8;
        const int f1 = k0 + 16 + half * 8;
        float av[16], bv[16];
        if (k0 + 32 <= N_FEAT_C) {
            // fast path: 16B-aligned float4 loads
            float4 xa0 = *(const float4*)(xrow + f0);
            float4 xa1 = *(const float4*)(xrow + f0 + 4);
            float4 xb0 = *(const float4*)(xrow + f1);
            float4 xb1 = *(const float4*)(xrow + f1 + 4);
            float4 wa0 = *(const float4*)(wrow + f0);
            float4 wa1 = *(const float4*)(wrow + f0 + 4);
            float4 wb0 = *(const float4*)(wrow + f1);
            float4 wb1 = *(const float4*)(wrow + f1 + 4);
            av[0]=xa0.x; av[1]=xa0.y; av[2]=xa0.z; av[3]=xa0.w;
            av[4]=xa1.x; av[5]=xa1.y; av[6]=xa1.z; av[7]=xa1.w;
            av[8]=xb0.x; av[9]=xb0.y; av[10]=xb0.z; av[11]=xb0.w;
            av[12]=xb1.x; av[13]=xb1.y; av[14]=xb1.z; av[15]=xb1.w;
            bv[0]=wa0.x*cmask; bv[1]=wa0.y*cmask; bv[2]=wa0.z*cmask; bv[3]=wa0.w*cmask;
            bv[4]=wa1.x*cmask; bv[5]=wa1.y*cmask; bv[6]=wa1.z*cmask; bv[7]=wa1.w*cmask;
            bv[8]=wb0.x*cmask; bv[9]=wb0.y*cmask; bv[10]=wb0.z*cmask; bv[11]=wb0.w*cmask;
            bv[12]=wb1.x*cmask; bv[13]=wb1.y*cmask; bv[14]=wb1.z*cmask; bv[15]=wb1.w*cmask;
        } else {
            // tail: 500..511 padded with zeros
            #pragma unroll
            for (int j = 0; j < 8; ++j) {
                int fA = f0 + j, fB = f1 + j;
                av[j]     = (fA < N_FEAT_C) ? xrow[fA] : 0.0f;
                av[8 + j] = (fB < N_FEAT_C) ? xrow[fB] : 0.0f;
                bv[j]     = (col_ok && fA < N_FEAT_C) ? wrow[fA] : 0.0f;
                bv[8 + j] = (col_ok && fB < N_FEAT_C) ? wrow[fB] : 0.0f;
            }
        }
        v16bf A, B;
        #pragma unroll
        for (int j = 0; j < 16; ++j) {
            A[j] = (__bf16)av[j];
            B[j] = (__bf16)bv[j];
        }
        acc = __builtin_amdgcn_wmma_f32_16x16x32_bf16(
            /*neg_a=*/false, A, /*neg_b=*/false, B,
            /*c_mod=*/(short)0, acc, /*reuse_a=*/false, /*reuse_b=*/false);
    }

    // C/D layout: VGPR r -> row (r + half*8), column lid. Store 8-wide padded rows.
    if (lid < 8) {
        #pragma unroll
        for (int r = 0; r < 8; ++r) {
            int node = node_base + r + half * 8;
            if (node < n_nodes) y[(size_t)node * 8 + lid] = acc[r];
        }
    }
}

// ---------------------------------------------------------------------------
// Propagation: y_next = dinv^2 * y_cur (self-loop, also initializes y_next),
// then scatter-add over edges: y_next[d] += dinv[s]*dinv[d]*y_cur[s]
// ---------------------------------------------------------------------------
__global__ __launch_bounds__(256)
void k_selfloop_init(const float* __restrict__ dinv, const float* __restrict__ yin,
                     float* __restrict__ yout, int n) {
    int i = blockIdx.x * blockDim.x + threadIdx.x;
    if (i >= n) return;
    float wgt = dinv[i] * dinv[i];
    const float4* pi = (const float4*)(yin + (size_t)i * 8);
    float4 a = pi[0], b = pi[1];
    float4* po = (float4*)(yout + (size_t)i * 8);
    po[0] = make_float4(wgt * a.x, wgt * a.y, wgt * a.z, wgt * a.w);
    po[1] = make_float4(wgt * b.x, wgt * b.y, wgt * b.z, wgt * b.w);
}

__global__ __launch_bounds__(256)
void k_edge_prop(const long long* __restrict__ src, const long long* __restrict__ dst,
                 const float* __restrict__ dinv, const float* __restrict__ yin,
                 float* __restrict__ yout, int e_cnt) {
    int e = blockIdx.x * blockDim.x + threadIdx.x;
    if (e >= e_cnt) return;
    int s = (int)src[e];
    int d = (int)dst[e];
    float wgt = dinv[s] * dinv[d];
    const float4* pi = (const float4*)(yin + (size_t)s * 8);
    float4 a = pi[0], b = pi[1];
    float* po = yout + (size_t)d * 8;
    atomicAdd(po + 0, wgt * a.x);
    atomicAdd(po + 1, wgt * a.y);
    atomicAdd(po + 2, wgt * a.z);
    atomicAdd(po + 3, wgt * a.w);
    atomicAdd(po + 4, wgt * b.x);
    atomicAdd(po + 5, wgt * b.y);
    atomicAdd(po + 6, wgt * b.z);
    // column 7 is identically zero (padding) -> skip its atomic
}

// ---------------------------------------------------------------------------
// Finalize: logits = y + bias; log_softmax over 7 classes
// ---------------------------------------------------------------------------
__global__ __launch_bounds__(256)
void k_finalize(const float* __restrict__ y, const float* __restrict__ bias,
                float* __restrict__ out, int n) {
    int i = blockIdx.x * blockDim.x + threadIdx.x;
    if (i >= n) return;
    const float* yr = y + (size_t)i * 8;
    float v[N_CLS_C];
    float m = -INFINITY;
    #pragma unroll
    for (int c = 0; c < N_CLS_C; ++c) {
        v[c] = yr[c] + bias[c];
        m = fmaxf(m, v[c]);
    }
    float s = 0.0f;
    #pragma unroll
    for (int c = 0; c < N_CLS_C; ++c) s += __expf(v[c] - m);
    float lse = __logf(s);
    #pragma unroll
    for (int c = 0; c < N_CLS_C; ++c) out[(size_t)i * N_CLS_C + c] = v[c] - m - lse;
}

// ---------------------------------------------------------------------------
extern "C" void kernel_launch(void* const* d_in, const int* in_sizes, int n_in,
                              void* d_out, int out_size, void* d_ws, size_t ws_size,
                              hipStream_t stream) {
    (void)n_in; (void)out_size; (void)ws_size;
    const float*     x    = (const float*)d_in[0];      // [N, 500]
    const float*     w    = (const float*)d_in[1];      // [7, 500]
    const float*     bias = (const float*)d_in[2];      // [7]
    const long long* ei   = (const long long*)d_in[3];  // [2, E] int64

    const int n = in_sizes[0] / N_FEAT_C;
    const int e_cnt = in_sizes[3] / 2;
    const long long* src = ei;          // row 0: message sources
    const long long* dst = ei + e_cnt;  // row 1: aggregation targets

    // Workspace layout (floats): dinv[n(pad4)] | y0[n*8] | y1[n*8]  (~6.1 MB)
    float* ws = (float*)d_ws;
    size_t np = (((size_t)n + 3) & ~(size_t)3);
    float* dinv = ws;
    float* y0 = ws + np;
    float* y1 = y0 + (size_t)n * 8;

    const int B = 256;
    const int gn = (n + B - 1) / B;
    const int ge = (e_cnt + B - 1) / B;
    const int n_tiles = (n + 15) / 16;
    const int gw = (n_tiles + 7) / 8;   // 8 waves per 256-thread block

    // deg -> dinv
    k_init_deg<<<gn, B, 0, stream>>>(dinv, n);
    k_count<<<ge, B, 0, stream>>>(dst, dinv, e_cnt);
    k_rsqrt_inplace<<<gn, B, 0, stream>>>(dinv, n);

    // project first (propagation is linear): y0 = X @ W^T  (padded N x 8)
    sgc_project_wmma<<<gw, B, 0, stream>>>(x, w, y0, n);

    // hop 1: y1 = A_hat y0
    k_selfloop_init<<<gn, B, 0, stream>>>(dinv, y0, y1, n);
    k_edge_prop<<<ge, B, 0, stream>>>(src, dst, dinv, y0, y1, e_cnt);

    // hop 2: y0 = A_hat y1
    k_selfloop_init<<<gn, B, 0, stream>>>(dinv, y1, y0, n);
    k_edge_prop<<<ge, B, 0, stream>>>(src, dst, dinv, y1, y0, e_cnt);

    // logits + log_softmax
    k_finalize<<<gn, B, 0, stream>>>(y0, bias, (float*)d_out, n);
}
